// WACE_78804059947542
// MI455X (gfx1250) — compile-verified, matrix operand order checked by reading, not run on hardware
//
#include <hip/hip_runtime.h>
#include <hip/hip_bf16.h>

typedef __bf16 bf16_t;
typedef bf16_t bf16x8 __attribute__((ext_vector_type(8)));
typedef bf16_t v16bf  __attribute__((ext_vector_type(16)));
typedef float  v8f    __attribute__((ext_vector_type(8)));

#define Bc   128
#define SPc  128
#define SHc  64
#define Hc   1024
#define H4c  4096
#define Ec   300
#define EPc  320
#define CDIV(a,b) (((a)+(b)-1)/(b))

__device__ inline float sigf(float x) { return 1.0f / (1.0f + __expf(-x)); }

// ---------------- WMMA fragment load: 16x32 bf16 tile slice for one lane ----
__device__ inline v16bf load_frag(const bf16_t* base, int ld, int row, int k0, int half) {
    const bf16_t* p = base + (size_t)row * ld + k0 + half * 8;
    bf16x8 lo = *(const bf16x8*)p;        // K = k0 + 8*half + [0..7]
    bf16x8 hi = *(const bf16x8*)(p + 16); // K = k0 + 16 + 8*half + [0..7]
    v16bf v;
    #pragma unroll
    for (int j = 0; j < 8; ++j) { v[j] = lo[j]; v[j + 8] = hi[j]; }
    return v;
}

struct Frags { v16bf a0, a1, b0, b1, b2, b3; };

__device__ inline Frags load_chunk(const bf16_t* A, const bf16_t* Bm, int K,
                                   int arow0, int arow1, int bcol0, int k0, int half) {
    Frags f;
    f.a0 = load_frag(A,  K, arow0,       k0, half);
    f.b0 = load_frag(Bm, K, bcol0,       k0, half);
    f.a1 = load_frag(A,  K, arow1,       k0, half);
    f.b1 = load_frag(Bm, K, bcol0 + 16,  k0, half);
    f.b2 = load_frag(Bm, K, bcol0 + 32,  k0, half);
    f.b3 = load_frag(Bm, K, bcol0 + 48,  k0, half);
    return f;
}

#define WMMA_BF16(a, b, c) \
    __builtin_amdgcn_wmma_f32_16x16x32_bf16(false, (a), false, (b), (short)0, (c), false, false)

__device__ inline void mma8(const Frags& f, v8f acc[2][4]) {
    acc[0][0] = WMMA_BF16(f.a0, f.b0, acc[0][0]);
    acc[1][0] = WMMA_BF16(f.a1, f.b0, acc[1][0]);
    acc[0][1] = WMMA_BF16(f.a0, f.b1, acc[0][1]);
    acc[1][1] = WMMA_BF16(f.a1, f.b1, acc[1][1]);
    acc[0][2] = WMMA_BF16(f.a0, f.b2, acc[0][2]);
    acc[1][2] = WMMA_BF16(f.a1, f.b2, acc[1][2]);
    acc[0][3] = WMMA_BF16(f.a0, f.b3, acc[0][3]);
    acc[1][3] = WMMA_BF16(f.a1, f.b3, acc[1][3]);
}

// one K-phase of the tile GEMM, double-buffered (K must be a multiple of 64)
__device__ inline void gemm_phase(const bf16_t* A, const bf16_t* Bm, int K,
                                  int arow0, int arow1, int bcol0, int half,
                                  v8f acc[2][4]) {
    Frags f0 = load_chunk(A, Bm, K, arow0, arow1, bcol0, 0, half);
    for (int k0 = 0; k0 < K; k0 += 64) {
        Frags f1 = load_chunk(A, Bm, K, arow0, arow1, bcol0, k0 + 32, half);
        mma8(f0, acc);
        if (k0 + 64 < K)
            f0 = load_chunk(A, Bm, K, arow0, arow1, bcol0, k0 + 64, half);
        mma8(f1, acc);
    }
}

// D[M,N] = A1[M,K1]*B1[N,K1]^T (+ A2[M,K2]*B2[N,K2]^T) (+ bias[N])
// one wave computes a 32x64 tile (2 m-frags x 4 n-frags = 8 wmma per K-chunk);
// block = 8 waves. obf: store D as bf16 instead of fp32.
// __launch_bounds__(256, 1): single wave per EU -> full register budget, no spills;
// latency is hidden by double-buffered fragment loads, not occupancy.
__global__ void __launch_bounds__(256, 1)
k_gemm_wmma(const bf16_t* A1, const bf16_t* B1, int K1,
            const bf16_t* A2, const bf16_t* B2, int K2,
            const float* bias, float* D, int M, int N, int obf) {
    int wave = threadIdx.x >> 5;
    int lane = threadIdx.x & 31;
    int tile = blockIdx.x * 8 + wave;
    int tiles_n = N >> 6;                     // 64-wide n tiles
    int tm = tile / tiles_n, tn = tile % tiles_n;
    int base_m = tm * 32, base_n = tn * 64;
    if (base_m >= M) return;
    int half = lane >> 4, r = lane & 15;
    int arow0 = base_m + r, arow1 = base_m + 16 + r;
    int bcol0 = base_n + r;
    v8f acc[2][4];
    #pragma unroll
    for (int i = 0; i < 2; ++i)
        #pragma unroll
        for (int j = 0; j < 4; ++j) acc[i][j] = (v8f){};

    gemm_phase(A1, B1, K1, arow0, arow1, bcol0, half, acc);
    if (A2)
        gemm_phase(A2, B2, K2, arow0, arow1, bcol0, half, acc);

    #pragma unroll
    for (int i = 0; i < 2; ++i) {
        #pragma unroll
        for (int j = 0; j < 4; ++j) {
            int col = base_n + j * 16 + r;
            float bs = bias ? bias[col] : 0.0f;
            size_t rowb = (size_t)(base_m + i * 16 + half * 8);
            if (obf) {
                bf16_t* dp = (bf16_t*)D + rowb * N + col;
                #pragma unroll
                for (int v = 0; v < 8; ++v) dp[(size_t)v * N] = (bf16_t)(acc[i][j][v] + bs);
            } else {
                float* dp = D + rowb * N + col;
                #pragma unroll
                for (int v = 0; v < 8; ++v) dp[(size_t)v * N] = acc[i][j][v] + bs;
            }
        }
    }
}

// ---------------- prep kernels ----------------------------------------------
__global__ void k_embed(const int* tok, const float* emb, bf16_t* dst,
                        int S, int Bn, int E, int Kpad) {
    int i = blockIdx.x * 256 + threadIdx.x;        // over S*Bn*Kpad, layout [t][b][Kpad]
    if (i >= S * Bn * Kpad) return;
    int e = i % Kpad; int sb = i / Kpad; int b = sb % Bn; int t = sb / Bn;
    float v = 0.0f;
    if (e < E) { int id = tok[b * S + t]; v = emb[(size_t)id * E + e]; }
    dst[i] = (bf16_t)v;
}

// convert fp32 weight to bf16 [N,Kpad]; trans: src is [Ksrc,N] else [N,Ksrc]
__global__ void k_conv(const float* src, bf16_t* dst, int N, int Ksrc, int Kpad, int trans) {
    int i = blockIdx.x * 256 + threadIdx.x;
    if (i >= N * Kpad) return;
    int k = i % Kpad; int n = i / Kpad;
    float v = 0.0f;
    if (k < Ksrc) v = trans ? src[(size_t)k * N + n] : src[(size_t)n * Ksrc + k];
    dst[i] = (bf16_t)v;
}

__global__ void k_addb(const float* a, const float* b, float* d, int n) {
    int i = blockIdx.x * 256 + threadIdx.x;
    if (i < n) d[i] = a[i] + b[i];
}

__global__ void k_zero(float* p, int n) {
    int i = blockIdx.x * 256 + threadIdx.x;
    if (i < n) p[i] = 0.0f;
}

__global__ void k_f2b(const float* s, bf16_t* d, int n) {
    int i = blockIdx.x * 256 + threadIdx.x;
    if (i < n) d[i] = (bf16_t)s[i];
}

// ---------------- LSTM cell + mask carry ------------------------------------
__global__ void k_lstm(const float* g, float* h, float* c, float* hm, float* cm,
                       bf16_t* h_bf, bf16_t* seq_out, size_t seq_stride_b,
                       const int* tok, int S, int t) {
    int i = blockIdx.x * 256 + threadIdx.x;     // over B*H
    if (i >= Bc * Hc) return;
    int b = i >> 10, hh = i & (Hc - 1);
    const float* gb = g + (size_t)b * H4c;
    float ig = sigf(gb[hh]);
    float fg = sigf(gb[Hc + hh]);
    float gg = tanhf(gb[2 * Hc + hh]);
    float og = sigf(gb[3 * Hc + hh]);
    float c2 = fg * c[i] + ig * gg;
    float h2 = og * tanhf(c2);
    c[i] = c2; h[i] = h2; h_bf[i] = (bf16_t)h2;
    float mt = (tok[b * S + t] != 0) ? 1.0f : 0.0f;
    float hm2 = mt * h2 + (1.0f - mt) * hm[i];
    hm[i] = hm2;
    if (cm) { cm[i] = mt * c2 + (1.0f - mt) * cm[i]; }
    seq_out[(size_t)b * seq_stride_b + hh] = (bf16_t)hm2;
}

// ---------------- attention kernels -----------------------------------------
// score[b,p] = dot(tanh(first[b,p,:] + second[b,:]), w) + (-1000 if pad)
// first is stored bf16 so the 64x re-read of [B,SP,H] stays resident in L2.
__global__ void k_score(const bf16_t* first, const float* second, const float* wv,
                        const int* prem, float* sc) {
    int bp = blockIdx.x;          // B*SP blocks
    int b = bp >> 7, p = bp & (SPc - 1);
    const bf16_t* f = first + (size_t)bp * Hc;
    const float* s = second + (size_t)b * Hc;
    float part = 0.0f;
    for (int hh = threadIdx.x; hh < Hc; hh += 256)
        part += tanhf((float)f[hh] + s[hh]) * wv[hh];
    __shared__ float red[256];
    red[threadIdx.x] = part; __syncthreads();
    for (int st = 128; st > 0; st >>= 1) {
        if (threadIdx.x < st) red[threadIdx.x] += red[threadIdx.x + st];
        __syncthreads();
    }
    if (threadIdx.x == 0)
        sc[bp] = red[0] + ((prem[b * SPc + p] != 0) ? 0.0f : -1000.0f);
}

__global__ void k_softmax128(float* sc) {       // one block (128 thr) per batch row
    int b = blockIdx.x, p = threadIdx.x;
    __shared__ float red[128];
    float v = sc[b * SPc + p];
    red[p] = v; __syncthreads();
    for (int st = 64; st > 0; st >>= 1) {
        if (p < st) red[p] = fmaxf(red[p], red[p + st]);
        __syncthreads();
    }
    float m = red[0]; __syncthreads();
    float e = __expf(v - m);
    red[p] = e; __syncthreads();
    for (int st = 64; st > 0; st >>= 1) {
        if (p < st) red[p] += red[p + st];
        __syncthreads();
    }
    sc[b * SPc + p] = e / red[0];
}

// aY[b,h] = sum_p alpha[b,p] * Y[b,p,h]   (Y bf16 [B,SP,H])
__global__ void k_alphaY(const float* alpha, const bf16_t* Ybf, float* aY) {
    int b = blockIdx.x >> 2;                    // H/256 = 4 chunks
    int hh = (blockIdx.x & 3) * 256 + threadIdx.x;
    __shared__ float al[SPc];
    if (threadIdx.x < SPc) al[threadIdx.x] = alpha[b * SPc + threadIdx.x];
    __syncthreads();
    const bf16_t* Yb = Ybf + (size_t)b * SPc * Hc + hh;
    float acc = 0.0f;
    #pragma unroll 4
    for (int p = 0; p < SPc; ++p) acc += al[p] * (float)Yb[(size_t)p * Hc];
    aY[b * Hc + hh] = acc;
}

__global__ void k_attupd(const float* aY, const float* rwt, float* r, bf16_t* r_bf,
                         const int* hyp, int t) {
    int i = blockIdx.x * 256 + threadIdx.x;
    if (i >= Bc * Hc) return;
    int b = i >> 10;
    float mt = (hyp[b * SHc + t] != 0) ? 1.0f : 0.0f;
    float rn = mt * (aY[i] + tanhf(rwt[i])) + (1.0f - mt) * r[i];
    r[i] = rn; r_bf[i] = (bf16_t)rn;
}

// tanh(hstar_pre) -> logits (C=4) -> softmax -> out
__global__ void k_final(const float* hstar_pre, const float* fcw, const float* fcb,
                        float* out) {
    int b = blockIdx.x, tid = threadIdx.x;      // 128 threads per block
    __shared__ float hs[Hc];
    for (int hh = tid; hh < Hc; hh += 128) hs[hh] = tanhf(hstar_pre[(size_t)b * Hc + hh]);
    __syncthreads();
    int cls = tid & 3;
    float part = 0.0f;
    for (int hh = (tid >> 2); hh < Hc; hh += 32) part += hs[hh] * fcw[(size_t)cls * Hc + hh];
    __shared__ float red[128];
    red[tid] = part; __syncthreads();
    if (tid < 4) {
        float s = 0.0f;
        for (int j = 0; j < 32; ++j) s += red[j * 4 + tid];
        red[tid] = s + fcb[tid];
    }
    __syncthreads();
    if (tid == 0) {
        float l0 = red[0], l1 = red[1], l2 = red[2], l3 = red[3];
        float m = fmaxf(fmaxf(l0, l1), fmaxf(l2, l3));
        float e0 = __expf(l0 - m), e1 = __expf(l1 - m);
        float e2 = __expf(l2 - m), e3 = __expf(l3 - m);
        float s = e0 + e1 + e2 + e3;
        out[b * 4 + 0] = e0 / s; out[b * 4 + 1] = e1 / s;
        out[b * 4 + 2] = e2 / s; out[b * 4 + 3] = e3 / s;
    }
}

// ---------------- host orchestration ----------------------------------------
extern "C" void kernel_launch(void* const* d_in, const int* in_sizes, int n_in,
                              void* d_out, int out_size, void* d_ws, size_t ws_size,
                              hipStream_t stream) {
    (void)in_sizes; (void)n_in; (void)out_size; (void)ws_size;
    const int*   premise    = (const int*)d_in[0];
    const int*   hypothesis = (const int*)d_in[1];
    const float* emb        = (const float*)d_in[2];
    const float* w_ih_p     = (const float*)d_in[3];
    const float* w_hh_p     = (const float*)d_in[4];
    const float* b_ih_p     = (const float*)d_in[5];
    const float* b_hh_p     = (const float*)d_in[6];
    const float* w_ih_h     = (const float*)d_in[7];
    const float* w_hh_h     = (const float*)d_in[8];
    const float* b_ih_h     = (const float*)d_in[9];
    const float* b_hh_h     = (const float*)d_in[10];
    const float* wy         = (const float*)d_in[11];
    const float* wh         = (const float*)d_in[12];
    const float* wr         = (const float*)d_in[13];
    const float* wv         = (const float*)d_in[14];
    const float* wt         = (const float*)d_in[15];
    const float* wp         = (const float*)d_in[16];
    const float* wx         = (const float*)d_in[17];
    const float* fcw        = (const float*)d_in[18];
    const float* fcb        = (const float*)d_in[19];
    float* out = (float*)d_out;

    char* ws = (char*)d_ws;
    size_t off = 0;
    auto alloc = [&](size_t bytes) -> void* {
        off = (off + 255) & ~(size_t)255;
        void* p = ws + off;
        off += bytes;
        return p;
    };
    bf16_t* x_p_bf    = (bf16_t*)alloc((size_t)SPc * Bc * EPc * 2);
    bf16_t* x_h_bf    = (bf16_t*)alloc((size_t)SHc * Bc * EPc * 2);
    bf16_t* w_ih_p_bf = (bf16_t*)alloc((size_t)H4c * EPc * 2);
    bf16_t* w_hh_p_bf = (bf16_t*)alloc((size_t)H4c * Hc * 2);
    bf16_t* w_ih_h_bf = (bf16_t*)alloc((size_t)H4c * EPc * 2);
    bf16_t* w_hh_h_bf = (bf16_t*)alloc((size_t)H4c * Hc * 2);
    bf16_t* wyT = (bf16_t*)alloc((size_t)Hc * Hc * 2);
    bf16_t* whT = (bf16_t*)alloc((size_t)Hc * Hc * 2);
    bf16_t* wrT = (bf16_t*)alloc((size_t)Hc * Hc * 2);
    bf16_t* wtT = (bf16_t*)alloc((size_t)Hc * Hc * 2);
    bf16_t* wpT = (bf16_t*)alloc((size_t)Hc * Hc * 2);
    bf16_t* wxT = (bf16_t*)alloc((size_t)Hc * Hc * 2);
    float* bias_p = (float*)alloc(H4c * 4);
    float* bias_h = (float*)alloc(H4c * 4);
    float* h  = (float*)alloc((size_t)Bc * Hc * 4);
    float* c  = (float*)alloc((size_t)Bc * Hc * 4);
    float* hm = (float*)alloc((size_t)Bc * Hc * 4);
    float* cm = (float*)alloc((size_t)Bc * Hc * 4);
    bf16_t* h_bf = (bf16_t*)alloc((size_t)Bc * Hc * 2);
    float* g = (float*)alloc((size_t)Bc * H4c * 4);
    bf16_t* Y_bf    = (bf16_t*)alloc((size_t)Bc * SPc * Hc * 2);   // [B,SP,H]
    bf16_t* outh_bf = (bf16_t*)alloc((size_t)SHc * Bc * Hc * 2);   // [SH,B,H]
    bf16_t* first_bf = (bf16_t*)alloc((size_t)Bc * SPc * Hc * 2);  // [B*SP,H] bf16
    float* second = (float*)alloc((size_t)Bc * Hc * 4);
    float* rwt    = (float*)alloc((size_t)Bc * Hc * 4);
    float* aY     = (float*)alloc((size_t)Bc * Hc * 4);
    float* r      = (float*)alloc((size_t)Bc * Hc * 4);
    bf16_t* r_bf  = (bf16_t*)alloc((size_t)Bc * Hc * 2);
    float* sc     = (float*)alloc((size_t)Bc * SPc * 4);
    float* hstar_pre = (float*)alloc((size_t)Bc * Hc * 4);
    bf16_t* hlast_bf = (bf16_t*)alloc((size_t)Bc * Hc * 2);

    auto gemm = [&](const bf16_t* A1, const bf16_t* B1, int K1,
                    const bf16_t* A2, const bf16_t* B2, int K2,
                    const float* bias, float* D, int M, int N, int obf) {
        int blocks = (M / 32) * (N / 64) / 8;
        k_gemm_wmma<<<blocks, 256, 0, stream>>>(A1, B1, K1, A2, B2, K2, bias, D, M, N, obf);
    };
    auto zero = [&](void* p, size_t bytes) {
        int n = (int)(bytes / 4);
        k_zero<<<CDIV(n, 256), 256, 0, stream>>>((float*)p, n);
    };

    // ---- prep: embeddings, bf16 weights, fused biases ----
    k_embed<<<CDIV(SPc * Bc * EPc, 256), 256, 0, stream>>>(premise, emb, x_p_bf, SPc, Bc, Ec, EPc);
    k_embed<<<CDIV(SHc * Bc * EPc, 256), 256, 0, stream>>>(hypothesis, emb, x_h_bf, SHc, Bc, Ec, EPc);
    k_conv<<<CDIV(H4c * EPc, 256), 256, 0, stream>>>(w_ih_p, w_ih_p_bf, H4c, Ec, EPc, 0);
    k_conv<<<CDIV(H4c * Hc, 256), 256, 0, stream>>>(w_hh_p, w_hh_p_bf, H4c, Hc, Hc, 0);
    k_conv<<<CDIV(H4c * EPc, 256), 256, 0, stream>>>(w_ih_h, w_ih_h_bf, H4c, Ec, EPc, 0);
    k_conv<<<CDIV(H4c * Hc, 256), 256, 0, stream>>>(w_hh_h, w_hh_h_bf, H4c, Hc, Hc, 0);
    k_conv<<<CDIV(Hc * Hc, 256), 256, 0, stream>>>(wy, wyT, Hc, Hc, Hc, 1);
    k_conv<<<CDIV(Hc * Hc, 256), 256, 0, stream>>>(wh, whT, Hc, Hc, Hc, 1);
    k_conv<<<CDIV(Hc * Hc, 256), 256, 0, stream>>>(wr, wrT, Hc, Hc, Hc, 1);
    k_conv<<<CDIV(Hc * Hc, 256), 256, 0, stream>>>(wt, wtT, Hc, Hc, Hc, 1);
    k_conv<<<CDIV(Hc * Hc, 256), 256, 0, stream>>>(wp, wpT, Hc, Hc, Hc, 1);
    k_conv<<<CDIV(Hc * Hc, 256), 256, 0, stream>>>(wx, wxT, Hc, Hc, Hc, 1);
    k_addb<<<CDIV(H4c, 256), 256, 0, stream>>>(b_ih_p, b_hh_p, bias_p, H4c);
    k_addb<<<CDIV(H4c, 256), 256, 0, stream>>>(b_ih_h, b_hh_h, bias_h, H4c);

    // ---- premise LSTM ----
    zero(h, (size_t)Bc * Hc * 4); zero(c, (size_t)Bc * Hc * 4);
    zero(hm, (size_t)Bc * Hc * 4); zero(cm, (size_t)Bc * Hc * 4);
    zero(h_bf, (size_t)Bc * Hc * 2);
    for (int t = 0; t < SPc; ++t) {
        gemm(h_bf, w_hh_p_bf, Hc, x_p_bf + (size_t)t * Bc * EPc, w_ih_p_bf, EPc,
             bias_p, g, Bc, H4c, 0);
        k_lstm<<<CDIV(Bc * Hc, 256), 256, 0, stream>>>(
            g, h, c, hm, cm, h_bf, Y_bf + (size_t)t * Hc, (size_t)SPc * Hc,
            premise, SPc, t);
    }

    // ---- hypothesis LSTM (h0 = 0, c0 = masked last premise cell) ----
    zero(h, (size_t)Bc * Hc * 4);
    hipMemcpyAsync(c, cm, (size_t)Bc * Hc * 4, hipMemcpyDeviceToDevice, stream);
    zero(hm, (size_t)Bc * Hc * 4);
    zero(h_bf, (size_t)Bc * Hc * 2);
    for (int t = 0; t < SHc; ++t) {
        gemm(h_bf, w_hh_h_bf, Hc, x_h_bf + (size_t)t * Bc * EPc, w_ih_h_bf, EPc,
             bias_h, g, Bc, H4c, 0);
        k_lstm<<<CDIV(Bc * Hc, 256), 256, 0, stream>>>(
            g, h, c, hm, (float*)nullptr, h_bf, outh_bf + (size_t)t * Bc * Hc, (size_t)Hc,
            hypothesis, SHc, t);
    }

    // ---- attention: first = Y @ wy (hoisted, stored bf16 to stay L2-resident) ----
    gemm(Y_bf, wyT, Hc, nullptr, nullptr, 0, nullptr, (float*)first_bf, Bc * SPc, Hc, 1);
    zero(r, (size_t)Bc * Hc * 4);
    zero(r_bf, (size_t)Bc * Hc * 2);
    for (int t = 0; t < SHc; ++t) {
        // second = h_t @ wh + r @ wr
        gemm(outh_bf + (size_t)t * Bc * Hc, whT, Hc, r_bf, wrT, Hc,
             nullptr, second, Bc, Hc, 0);
        // rwt = r @ wt
        gemm(r_bf, wtT, Hc, nullptr, nullptr, 0, nullptr, rwt, Bc, Hc, 0);
        k_score<<<Bc * SPc, 256, 0, stream>>>(first_bf, second, wv, premise, sc);
        k_softmax128<<<Bc, 128, 0, stream>>>(sc);
        k_alphaY<<<Bc * (Hc / 256), 256, 0, stream>>>(sc, Y_bf, aY);
        k_attupd<<<CDIV(Bc * Hc, 256), 256, 0, stream>>>(aY, rwt, r, r_bf, hypothesis, t);
    }

    // ---- h_star = tanh(r @ wp + out_h_last @ wx); logits; softmax ----
    k_f2b<<<CDIV(Bc * Hc, 256), 256, 0, stream>>>(hm, hlast_bf, Bc * Hc);
    gemm(r_bf, wpT, Hc, hlast_bf, wxT, Hc, nullptr, hstar_pre, Bc, Hc, 0);
    k_final<<<Bc, 128, 0, stream>>>(hstar_pre, fcw, fcb, out);
}
